// MyNER_27565100105862
// MI455X (gfx1250) — matrix operand order repaired and checked by reading, not compiled, not run on hardware
//
#include <hip/hip_runtime.h>
#include <hip/hip_bf16.h>
#include <math.h>

// ---------- problem constants ----------
#define BB   64
#define SS   256
#define EE   256
#define HH   256
#define GH   1024          // 4*H
#define LL   9
#define ROWS (BB*SS)       // 16384

typedef _Float16 v16h __attribute__((ext_vector_type(16)));
typedef _Float16 v8h  __attribute__((ext_vector_type(8)));
typedef float    v8f  __attribute__((ext_vector_type(8)));

// ---------- workspace layout (bytes) ----------
// x_f16      : ROWS*EE*2                = 8 MB
// Wt (4 mats): 4 * GH*EE*2              = 2 MB   (W_fwdT, W_bwdT, U_fwdT, U_bwdT)
// xz (2 dirs): 2 * ROWS*GH*2            = 64 MB
// h  (2 dirs): 2 * ROWS*HH*2            = 16 MB
#define OFF_X16   0
#define OFF_WT    (OFF_X16 + (size_t)ROWS*EE*2)
#define OFF_XZ    (OFF_WT  + (size_t)4*GH*EE*2)
#define OFF_H     (OFF_XZ  + (size_t)2*ROWS*GH*2)

// ---------------------------------------------------------------------------
// Fragment loaders per CDNA5 ISA 7.12.2 (wave32, 16-bit 16x32 A / 32x16 B)
// A: lane<16 -> row=lane, K chunks [k0..k0+7],[k0+16..k0+23]
//    lane>=16 -> same row, K chunks [k0+8..k0+15],[k0+24..k0+31]
__device__ __forceinline__ v16h load_frag_a(const _Float16* __restrict__ base,
                                            int stride, int lane, int k0) {
    int r   = lane & 15;
    int sel = (lane >> 4) & 1;
    const _Float16* p = base + (size_t)r * stride + k0 + sel * 8;
    union { v16h v; v8h h[2]; } u;
    u.h[0] = *(const v8h*)(p);
    u.h[1] = *(const v8h*)(p + 16);
    return u.v;
}
// B (stored as N x K row-major, i.e. W^T): lane<16 -> col=lane, K [k0..k0+7],[k0+16..k0+23]
//    lane>=16 -> same col, K [k0+8..k0+15],[k0+24..k0+31]   (same order as A)
__device__ __forceinline__ v16h load_frag_b(const _Float16* __restrict__ base,
                                            int stride, int lane, int k0) {
    int n   = lane & 15;
    int sel = (lane >> 4) & 1;
    const _Float16* p = base + (size_t)n * stride + k0 + sel * 8;
    union { v16h v; v8h h[2]; } u;
    u.h[0] = *(const v8h*)(p);
    u.h[1] = *(const v8h*)(p + 16);
    return u.v;
}

__device__ __forceinline__ float sigm(float x) { return 1.0f / (1.0f + expf(-x)); }

// 32-bit LDS byte offset of a generic pointer that points into LDS
__device__ __forceinline__ unsigned lds_addr_u32(const void* p) {
    return (unsigned)(size_t)(__attribute__((address_space(3))) const void*)p;
}

// ---------------------------------------------------------------------------
// K0: convert W_fwd, W_bwd, U_fwd, U_bwd (E x 4H f32) -> f16 transposed (4H x E)
__global__ void wt_prep_kernel(const float* __restrict__ Wf, const float* __restrict__ Wb,
                               const float* __restrict__ Uf, const float* __restrict__ Ub,
                               _Float16* __restrict__ Wt) {
    int idx = blockIdx.x * blockDim.x + threadIdx.x;      // 0 .. 4*GH*EE-1
    int m   = idx >> 18;                                  // GH*EE = 262144 = 2^18
    int rem = idx & (GH * EE - 1);
    int n   = rem >> 8;                                   // 0..1023
    int k   = rem & 255;                                  // 0..255
    const float* src = (m == 0) ? Wf : (m == 1) ? Wb : (m == 2) ? Uf : Ub;
    Wt[(size_t)m * GH * EE + (size_t)n * EE + k] = (_Float16)src[(size_t)k * GH + n];
}

// K1: embedding gather -> f16
__global__ void embed_kernel(const int* __restrict__ ids, const float* __restrict__ emb,
                             _Float16* __restrict__ x16) {
    int row = blockIdx.x;                 // 0..ROWS-1
    int e   = threadIdx.x;                // 0..255
    x16[(size_t)row * EE + e] = (_Float16)emb[(size_t)ids[row] * EE + e];
}

// K2: xz = x @ W + b, both directions, f16 in / f32 acc / f16 out.
// Each wave: 32x64 output tile (2 M-tiles x 4 N-tiles), K loop of 8.
__global__ void xz_gemm_kernel(const _Float16* __restrict__ x16,
                               const _Float16* __restrict__ Wt_all,
                               const float* __restrict__ b_fwd,
                               const float* __restrict__ b_bwd,
                               _Float16* __restrict__ xz_all) {
    int dir = blockIdx.y;
    const _Float16* Wt   = Wt_all + (size_t)dir * GH * EE;         // W_fwdT / W_bwdT
    const float*    bias = dir ? b_bwd : b_fwd;
    _Float16*       xz   = xz_all + (size_t)dir * ROWS * GH;

    int lane = threadIdx.x & 31;
    int wave = (blockIdx.x * blockDim.x + threadIdx.x) >> 5;  // global wave id
    int m0 = (wave >> 4) * 32;   // 512 M-groups of 32 rows
    int n0 = (wave & 15) * 64;   // 16 N-groups of 64 cols

    v8f acc[2][4] = {};
    for (int k0 = 0; k0 < EE; k0 += 32) {
        v16h a0 = load_frag_a(x16 + (size_t)m0 * EE, EE, lane, k0);
        v16h a1 = load_frag_a(x16 + (size_t)(m0 + 16) * EE, EE, lane, k0);
#pragma unroll
        for (int nn = 0; nn < 4; ++nn) {
            v16h b = load_frag_b(Wt + (size_t)(n0 + 16 * nn) * EE, EE, lane, k0);
            acc[0][nn] = __builtin_amdgcn_wmma_f32_16x16x32_f16(
                false, a0, false, b, (short)0, acc[0][nn], false, false);
            acc[1][nn] = __builtin_amdgcn_wmma_f32_16x16x32_f16(
                false, a1, false, b, (short)0, acc[1][nn], false, false);
        }
    }
    // D layout: VGPR r, lanes 0-15 -> row r, col lane; lanes 16-31 -> row 8+r
    int col_l = lane & 15;
    int row_l = (lane >> 4) * 8;
#pragma unroll
    for (int mi = 0; mi < 2; ++mi)
#pragma unroll
        for (int nn = 0; nn < 4; ++nn) {
            int n  = n0 + 16 * nn + col_l;
            float bv = bias[n];
#pragma unroll
            for (int r = 0; r < 8; ++r) {
                int row = m0 + 16 * mi + row_l + r;
                xz[(size_t)row * GH + n] = (_Float16)(acc[mi][nn][r] + bv);
            }
        }
}

// K3: sequential LSTM scan, one 1024-thread workgroup per direction.
// LDS: h (64x256 f16, 32KB) + z (64x1024 f16, 128KB) + xz stage (64x1024 f16,
// 128KB) = 288KB dynamic (<= 320KB/WGP on CDNA5).
// Per step, each wave async-copies its own 4KB xz slice to LDS
// (GLOBAL_LOAD_ASYNC_TO_LDS_B128, ASYNCcnt) and overlaps it with the 64-WMMA
// h@U GEMM, then waits with s_wait_asynccnt before the epilogue.
__global__ void __launch_bounds__(1024, 1)
lstm_scan_kernel(const _Float16* __restrict__ xz_all,
                 const _Float16* __restrict__ Ut_all,
                 _Float16* __restrict__ h_all) {
    extern __shared__ char smem[];
    _Float16* h_lds  = (_Float16*)smem;              // [64][256]
    _Float16* z_lds  = h_lds + BB * HH;              // [64][1024]
    _Float16* xzs    = z_lds + BB * GH;              // [64][1024] async stage

    int dir = blockIdx.x;
    const _Float16* xz  = xz_all + (size_t)dir * ROWS * GH;
    const _Float16* Ut  = Ut_all + (size_t)dir * GH * EE;   // U_fwdT / U_bwdT
    _Float16*       hout = h_all + (size_t)dir * ROWS * HH;

    int tid = threadIdx.x;
    for (int i = tid; i < BB * HH; i += 1024) h_lds[i] = (_Float16)0.0f;
    float c[16];
#pragma unroll
    for (int r = 0; r < 16; ++r) c[r] = 0.0f;
    __syncthreads();

    int lane  = tid & 31;
    int wave  = tid >> 5;            // 0..31
    int m0    = (wave >> 3) * 16;    // 4 row-groups of 16 (batch)
    int n0    = (wave & 7) * 128;    // 8 col-groups of 128
    int col_l = lane & 15;
    int row_l = (lane >> 4) * 8;

    unsigned xzs_base = lds_addr_u32(xzs);

    for (int t = 0; t < SS; ++t) {
        int te = dir ? (SS - 1 - t) : t;

        // ---- issue async copy of this wave's xz slice (rows m0..m0+15,
        //      cols n0..n0+127): 256 x 16B chunks, 8 per lane ----
#pragma unroll
        for (int i = 0; i < 8; ++i) {
            int chunk = i * 32 + lane;             // 0..255
            int row   = m0 + (chunk >> 4);         // 16 chunks of 16B per row
            int boff  = (chunk & 15) * 16;         // byte offset in 256B row slice
            const char* g = (const char*)xz
                          + (((size_t)row * SS + te) * GH + n0) * 2 + boff;
            unsigned l = xzs_base + (unsigned)((row * GH + n0) * 2 + boff);
            asm volatile("global_load_async_to_lds_b128 %0, %1, off"
                         :: "v"(l), "v"(g) : "memory");
        }

        // ---- z-slice = h @ U  (16 x 128 per wave, K=256) ----
        v8f acc[8] = {};
        for (int k0 = 0; k0 < HH; k0 += 32) {
            v16h a = load_frag_a(h_lds + (size_t)m0 * HH, HH, lane, k0);
#pragma unroll
            for (int nn = 0; nn < 8; ++nn) {
                v16h b = load_frag_b(Ut + (size_t)(n0 + 16 * nn) * EE, EE, lane, k0);
                acc[nn] = __builtin_amdgcn_wmma_f32_16x16x32_f16(
                    false, a, false, b, (short)0, acc[nn], false, false);
            }
        }

        // ---- wait for this wave's async xz slice, then z += xz, stash z ----
        asm volatile("s_wait_asynccnt 0x0" ::: "memory");
#pragma unroll
        for (int nn = 0; nn < 8; ++nn) {
            int n = n0 + 16 * nn + col_l;
#pragma unroll
            for (int r = 0; r < 8; ++r) {
                int b = m0 + row_l + r;   // batch row
                float z = acc[nn][r] + (float)xzs[b * GH + n];
                z_lds[b * GH + n] = (_Float16)z;
            }
        }
        __syncthreads();

        // ---- elementwise gates: 16384 (b,j) pairs / 1024 threads = 16 each ----
#pragma unroll
        for (int r = 0; r < 16; ++r) {
            int idx = tid + (r << 10);
            int b = idx >> 8;
            int j = idx & 255;
            float zi = (float)z_lds[b * GH + j];
            float zf = (float)z_lds[b * GH + HH + j];
            float zg = (float)z_lds[b * GH + 2 * HH + j];
            float zo = (float)z_lds[b * GH + 3 * HH + j];
            float cn = sigm(zf) * c[r] + sigm(zi) * tanhf(zg);
            float hn = sigm(zo) * tanhf(cn);
            c[r] = cn;
            h_lds[b * HH + j] = (_Float16)hn;
            hout[((size_t)b * SS + te) * HH + j] = (_Float16)hn;
        }
        __syncthreads();
    }
}

// K4: logits = [h_f ; h_b] @ W_dense + b_dense  (512 -> 9), W_dense staged in LDS
__global__ void dense_kernel(const _Float16* __restrict__ h_all,
                             const float* __restrict__ Wd,
                             const float* __restrict__ bd,
                             float* __restrict__ logits) {
    __shared__ float wds[2 * HH * LL];
    int tid = threadIdx.x;
    for (int i = tid; i < 2 * HH * LL; i += blockDim.x) wds[i] = Wd[i];
    __syncthreads();

    int i = blockIdx.x * blockDim.x + tid;   // 0..ROWS-1
    const _Float16* hf = h_all + (size_t)i * HH;
    const _Float16* hb = h_all + (size_t)ROWS * HH + (size_t)i * HH;
    float acc[LL];
#pragma unroll
    for (int l = 0; l < LL; ++l) acc[l] = bd[l];
    for (int k = 0; k < HH; ++k) {
        float hv = (float)hf[k];
#pragma unroll
        for (int l = 0; l < LL; ++l) acc[l] += hv * wds[k * LL + l];
    }
    for (int k = 0; k < HH; ++k) {
        float hv = (float)hb[k];
#pragma unroll
        for (int l = 0; l < LL; ++l) acc[l] += hv * wds[(HH + k) * LL + l];
    }
#pragma unroll
    for (int l = 0; l < LL; ++l) logits[(size_t)i * LL + l] = acc[l];
}

// K5: CRF log-likelihood per batch (thread b), plus trans passthrough
__global__ void crf_kernel(const float* __restrict__ logits,
                           const int* __restrict__ label,
                           const int* __restrict__ mask,
                           const float* __restrict__ trans,
                           float* __restrict__ ll_out,
                           float* __restrict__ trans_out) {
    __shared__ float tr[LL * LL];
    int tid = threadIdx.x;
    if (tid < LL * LL) { tr[tid] = trans[tid]; trans_out[tid] = trans[tid]; }
    __syncthreads();
    if (tid >= BB) return;

    int b = tid;
    const float* lg = logits + (size_t)b * SS * LL;
    const int*   tg = label + (size_t)b * SS;

    int len = 0;
    for (int s = 0; s < SS; ++s) len += (mask[(size_t)b * SS + s] != 0);

    float unary = 0.0f;
    for (int s = 0; s < len; ++s) unary += lg[s * LL + tg[s]];
    float binary = 0.0f;
    for (int s = 0; s + 1 < len; ++s) binary += tr[tg[s] * LL + tg[s + 1]];

    float alpha[LL];
#pragma unroll
    for (int j = 0; j < LL; ++j) alpha[j] = lg[j];
    for (int s = 1; s < SS; ++s) {
        if (s < len) {
            float na[LL];
            for (int j = 0; j < LL; ++j) {
                float m = -1e30f;
                for (int i2 = 0; i2 < LL; ++i2) {
                    float v = alpha[i2] + tr[i2 * LL + j];
                    m = v > m ? v : m;
                }
                float ss = 0.0f;
                for (int i2 = 0; i2 < LL; ++i2)
                    ss += expf(alpha[i2] + tr[i2 * LL + j] - m);
                na[j] = m + logf(ss) + lg[s * LL + j];
            }
#pragma unroll
            for (int j = 0; j < LL; ++j) alpha[j] = na[j];
        }
    }
    float m = -1e30f;
#pragma unroll
    for (int j = 0; j < LL; ++j) m = alpha[j] > m ? alpha[j] : m;
    float ss = 0.0f;
#pragma unroll
    for (int j = 0; j < LL; ++j) ss += expf(alpha[j] - m);
    ll_out[b] = unary + binary - (m + logf(ss));
}

// ---------------------------------------------------------------------------
extern "C" void kernel_launch(void* const* d_in, const int* in_sizes, int n_in,
                              void* d_out, int out_size, void* d_ws, size_t ws_size,
                              hipStream_t stream) {
    const int*   inputs  = (const int*)d_in[0];
    const int*   amask   = (const int*)d_in[1];
    const int*   label   = (const int*)d_in[2];
    const float* emb     = (const float*)d_in[3];
    const float* W_fwd   = (const float*)d_in[4];
    const float* U_fwd   = (const float*)d_in[5];
    const float* b_fwd   = (const float*)d_in[6];
    const float* W_bwd   = (const float*)d_in[7];
    const float* U_bwd   = (const float*)d_in[8];
    const float* b_bwd   = (const float*)d_in[9];
    const float* W_dense = (const float*)d_in[10];
    const float* b_dense = (const float*)d_in[11];
    const float* trans   = (const float*)d_in[12];

    char* ws = (char*)d_ws;
    _Float16* x16 = (_Float16*)(ws + OFF_X16);
    _Float16* Wt  = (_Float16*)(ws + OFF_WT);
    _Float16* xz  = (_Float16*)(ws + OFF_XZ);
    _Float16* hbf = (_Float16*)(ws + OFF_H);
    float* out = (float*)d_out;

    // K0: weight transpose/convert (4 * 1024*256 elems)
    wt_prep_kernel<<<4 * GH * EE / 256, 256, 0, stream>>>(W_fwd, W_bwd, U_fwd, U_bwd, Wt);
    // K1: embedding gather
    embed_kernel<<<ROWS, EE, 0, stream>>>(inputs, emb, x16);
    // K2: input projections (both directions)
    dim3 g2(ROWS / 32 * (GH / 64) / 8, 2);   // 8 waves per 256-thread block
    xz_gemm_kernel<<<g2, 256, 0, stream>>>(x16, Wt, b_fwd, b_bwd, xz);
    // K3: recurrent scan — 2 workgroups (fwd/bwd), 288KB dynamic LDS each
    size_t lds_bytes = (size_t)(BB * HH + 2 * BB * GH) * sizeof(_Float16);
    lstm_scan_kernel<<<2, 1024, lds_bytes, stream>>>(xz, Wt + (size_t)2 * GH * EE, hbf);
    // K4: dense logits
    dense_kernel<<<ROWS / 256, 256, 0, stream>>>(hbf, W_dense, b_dense, out);
    // K5: CRF + trans passthrough
    crf_kernel<<<1, 128, 0, stream>>>(out, label, amask, trans,
                                      out + (size_t)ROWS * LL,
                                      out + (size_t)ROWS * LL + BB);
}
